// SimplePAC_55868934586707
// MI455X (gfx1250) — matrix operand order repaired and checked by reading, compiled, not run on hardware
//
#include <hip/hip_runtime.h>
#include <math.h>

// ---------------------------------------------------------------------------
// PAC (phase-amplitude coupling) for MI455X / gfx1250.
//   Stage 0: build f16 WMMA filter banks, pre-swizzled into A-fragment order
//            (real bandpass from inputs + closed-form quadrature partner).
//   Stage 1: WMMA FIR filtfilt + analytic signal, fused phase-bin / amplitude.
//            B operands built with dword-aligned LDS loads + v_alignbit.
//   Stage 2: deterministic LDS-histogram Tort modulation index.
// Workspace requirement: ~26.3 MB.
// ---------------------------------------------------------------------------

typedef __attribute__((ext_vector_type(16))) _Float16 v16h;
typedef __attribute__((ext_vector_type(8)))  float    v8f;
typedef __attribute__((ext_vector_type(8)))  unsigned int v8u;

#define FS_F     512.0f
#define T_LEN    4096
#define N_SIG    128          // 4*8*4 signals
#define NB       10           // bands per bank
#define N_BINS   18
#define L_PHA    385
#define L_AMP    193
#define C_PHA    192          // (L-1)/2
#define C_AMP    96
#define CH_PHA   13           // ceil(385/32)
#define CH_AMP   7            // ceil(193/32)

// workspace layout (bytes); swizzled bank = chunks*32 lanes*16 halves
#define WS_PHA_R 0
#define WS_PHA_Q (WS_PHA_R + CH_PHA*32*16*2)         // 13312
#define WS_AMP_R (WS_PHA_Q + CH_PHA*32*16*2)         // 26624
#define WS_AMP_Q (WS_AMP_R + CH_AMP*32*16*2)         // 33792
#define WS_BINS  40960                               // u8 [128][10][4096]
#define WS_AMPS  (WS_BINS + N_SIG*NB*T_LEN)          // f32 [128][10][4096]

// ---------------------------------------------------------------------------
// Stage 0: f16 filter banks in WMMA A-fragment order.
// Fragment element e of (chunk kc, lane l) holds tap
//   row = l&15,  col = kc*32 + (l>>4)*16 + e
// (contiguous k order, same as the B fragments).  Rows >= 10 and cols >= L are zero so A-tiles need
// no masking.  Quadrature bank = analytic partner of the windowed-sinc BP:
//   h (τ) = [sin(2π f2 τ/fs) - sin(2π f1 τ/fs)]/(π τ) * w(τ)
//   hq(τ) = [cos(2π f1 τ/fs) - cos(2π f2 τ/fs)]/(π τ) * w(τ)   (hq(0)=0)
// ---------------------------------------------------------------------------
__global__ void pac_prep_banks(const float* __restrict__ phaK,
                               const float* __restrict__ ampK,
                               _Float16* __restrict__ phaR, _Float16* __restrict__ phaQ,
                               _Float16* __restrict__ ampR, _Float16* __restrict__ ampQ) {
    const float PI = 3.14159265358979f;
    int id = blockIdx.x * blockDim.x + threadIdx.x;

    if (id < CH_PHA * 32 * 16) {                    // phase bank: (2,4)..(20,22)
        int kc = id >> 9, rem = id & 511;
        int lane = rem >> 4, e = rem & 15;
        int row = lane & 15;
        int col = kc * 32 + ((lane >> 4) * 16) + e;
        float hr = 0.f, hq = 0.f;
        if (row < NB && col < L_PHA) {
            hr = phaK[row * L_PHA + col];
            float f1 = 2.0f + 2.0f * (float)row, f2 = f1 + 2.0f;
            float tau = (float)col - (float)C_PHA;
            float ham = 0.54f - 0.46f * cosf(2.f * PI * (float)col / (float)(L_PHA - 1));
            if (tau != 0.f)
                hq = (cosf(2.f * PI * f1 / FS_F * tau) -
                      cosf(2.f * PI * f2 / FS_F * tau)) / (PI * tau) * ham;
        }
        phaR[id] = (_Float16)hr;
        phaQ[id] = (_Float16)hq;
        return;
    }
    int id2 = id - CH_PHA * 32 * 16;
    if (id2 < CH_AMP * 32 * 16) {                   // amplitude bank: (30,45)..(165,180)
        int kc = id2 >> 9, rem = id2 & 511;
        int lane = rem >> 4, e = rem & 15;
        int row = lane & 15;
        int col = kc * 32 + ((lane >> 4) * 16) + e;
        float hr = 0.f, hq = 0.f;
        if (row < NB && col < L_AMP) {
            hr = ampK[row * L_AMP + col];
            float f1 = 30.0f + 15.0f * (float)row, f2 = f1 + 15.0f;
            float tau = (float)col - (float)C_AMP;
            float ham = 0.54f - 0.46f * cosf(2.f * PI * (float)col / (float)(L_AMP - 1));
            if (tau != 0.f)
                hq = (cosf(2.f * PI * f1 / FS_F * tau) -
                      cosf(2.f * PI * f2 / FS_F * tau)) / (PI * tau) * ham;
        }
        ampR[id2] = (_Float16)hr;
        ampQ[id2] = (_Float16)hq;
    }
}

// ---------------------------------------------------------------------------
// Stage 1 helpers
// ---------------------------------------------------------------------------
__device__ __forceinline__ unsigned short pac_half_at(const float* __restrict__ xrow,
                                                      int i, int rsz, int tblock, int ctr) {
    if ((unsigned)i >= (unsigned)rsz) return 0;     // zero-fill pad region
    int g = tblock - ctr + i;
    if (g < 0)      g = -g;                         // jnp 'reflect' (no edge dup)
    if (g >= T_LEN) g = 2 * T_LEN - 2 - g;
    _Float16 h = (_Float16)xrow[g];
    return __builtin_bit_cast(unsigned short, h);
}

// ---------------------------------------------------------------------------
// Stage 1: WMMA FIR filtfilt + analytic signal + fused bin/amplitude emit.
// Block = 256 threads = 8 waves; wave w owns output tile [tblock+16w, +16).
// Window sx[i] = xr[tblock - ctr + i] staged in LDS twice, packed 2 halves per
// dword: sxF (forward order) and sxR (reversed, sxR[j] = sx[575-j]) so both
// the fwd fragment (xr[t0+N+k-c]) and the filtfilt bwd fragment (xr[t0+N-k+c])
// are forward-contiguous 16-half runs per lane.  Start parity per lane is
// chunk-invariant, so each fragment = 9 aligned dword LDS loads + 8 alignbits.
//   re = 0.5*(corr_fwd + corr_bwd)  (symmetric real kernel)
//   im = corr_fwd with quadrature kernel (antisymmetric -> bwd would cancel)
// ---------------------------------------------------------------------------
template <int BANK>
__global__ void __launch_bounds__(256)
pac_filter_wmma(const float* __restrict__ x,
                const _Float16* __restrict__ phaR, const _Float16* __restrict__ phaQ,
                const _Float16* __restrict__ ampR, const _Float16* __restrict__ ampQ,
                unsigned char* __restrict__ bins, float* __restrict__ amps) {
    constexpr int L   = BANK ? L_AMP : L_PHA;
    constexpr int CTR = BANK ? C_AMP : C_PHA;
    constexpr int NCH = BANK ? CH_AMP : CH_PHA;
    constexpr int RSZ = 128 + L - 1;                // 512 / 320 valid halves

    const int n      = blockIdx.y;                  // signal index 0..127
    const int tblock = blockIdx.x * 128;            // 128 output samples / block
    const float* __restrict__ xrow = x + n * T_LEN;
    const v16h* __restrict__ fragR = (const v16h*)(BANK ? ampR : phaR);
    const v16h* __restrict__ fragQ = (const v16h*)(BANK ? ampQ : phaQ);

    // 640 halves = 320 dwords per copy (covers padded-chunk overreach, zeroed)
    __shared__ unsigned int sxF[320];
    __shared__ unsigned int sxR[320];
    for (int wi = threadIdx.x; wi < 320; wi += 256) {
        int i0 = 2 * wi, i1 = 2 * wi + 1;
        unsigned int f0 = pac_half_at(xrow, i0,       RSZ, tblock, CTR);
        unsigned int f1 = pac_half_at(xrow, i1,       RSZ, tblock, CTR);
        unsigned int r0 = pac_half_at(xrow, 575 - i0, RSZ, tblock, CTR);
        unsigned int r1 = pac_half_at(xrow, 575 - i1, RSZ, tblock, CTR);
        sxF[wi] = f0 | (f1 << 16);
        sxR[wi] = r0 | (r1 << 16);
    }
    __syncthreads();

    const int wave = threadIdx.x >> 5;
    const int lane = threadIdx.x & 31;
    const int hi   = lane >> 4;
    const int lo   = lane & 15;
    const int tloc = wave * 16;

    // fragment start half-indices (chunk adds klo = kc*32 + hi*16, always even)
    const int sF0 = tloc + lo;                      // fwd start in sxF
    const int sB0 = 575 - tloc - lo - 2 * CTR;      // bwd start in sxR
    const unsigned int shF = (unsigned)(sF0 & 1) * 16u;
    const unsigned int shB = (unsigned)(sB0 & 1) * 16u;

    v8f cR = {};                                    // real accumulator (fwd+bwd)
    v8f cI = {};                                    // quadrature accumulator (fwd)
#pragma unroll
    for (int kc = 0; kc < NCH; ++kc) {
        const int klo   = kc * 32 + hi * 16;
        const int baseF = (sF0 + klo) >> 1;
        const int baseB = (sB0 + klo) >> 1;
        v8u oF, oB;
#pragma unroll
        for (int i = 0; i < 8; ++i) {
            oF[i] = __builtin_amdgcn_alignbit(sxF[baseF + i + 1], sxF[baseF + i], shF);
            oB[i] = __builtin_amdgcn_alignbit(sxR[baseB + i + 1], sxR[baseB + i], shB);
        }
        v16h bF = __builtin_bit_cast(v16h, oF);
        v16h bB = __builtin_bit_cast(v16h, oB);
        v16h aR = fragR[kc * 32 + lane];            // 2x global_load_b128, WGP$-hot
        v16h aQ = fragQ[kc * 32 + lane];
        cR = __builtin_amdgcn_wmma_f32_16x16x32_f16(false, aR, false, bF, (short)0, cR, false, false);
        cR = __builtin_amdgcn_wmma_f32_16x16x32_f16(false, aR, false, bB, (short)0, cR, false, false);
        cI = __builtin_amdgcn_wmma_f32_16x16x32_f16(false, aQ, false, bF, (short)0, cI, false, false);
    }

    // C/D f32 layout: VGPR v, lane -> M = v + hi*8 (filter row), N = lo (time)
    const float PI    = 3.14159265358979f;
    const float width = 2.f * PI / (float)N_BINS;
#pragma unroll
    for (int v = 0; v < 8; ++v) {
        int f = v + hi * 8;
        if (f < NB) {
            int t    = tblock + tloc + lo;
            float re = 0.5f * cR[v];
            float im = cI[v];
            if (BANK == 0) {
                float ph = atan2f(im, re);
                int b = (int)floorf((ph + PI) / width);
                b = b < 0 ? 0 : (b > N_BINS - 1 ? N_BINS - 1 : b);
                bins[(n * NB + f) * T_LEN + t] = (unsigned char)b;
            } else {
                amps[(n * NB + f) * T_LEN + t] = sqrtf(re * re + im * im);
            }
        }
    }
}

// ---------------------------------------------------------------------------
// Stage 2: Tort MI.  One block per output (b,c,p',a').  Per-thread privatized
// LDS histograms (no float atomics -> replay-deterministic), fixed-order
// reduction, entropy on thread 0.  Replicates the reference's (B*C*S,P,T) ->
// (B,C,P,S,T) reshape scramble via q = p'*4+s' -> (s=q/10, p=q%10).
// ---------------------------------------------------------------------------
__global__ void __launch_bounds__(256)
pac_mi(const unsigned char* __restrict__ bins,
       const float* __restrict__ amps,
       float* __restrict__ out) {
    const int aP  = blockIdx.x;                     // a' 0..9
    const int pP  = blockIdx.y;                     // p' 0..9
    const int bc  = blockIdx.z;                     // (b*8+c) 0..31
    const int tid = threadIdx.x;

    __shared__ float hsum[256 * N_BINS];
    __shared__ float hcnt[256 * N_BINS];
    for (int k = 0; k < N_BINS; ++k) {
        hsum[tid * N_BINS + k] = 0.f;
        hcnt[tid * N_BINS + k] = 0.f;
    }

    for (int s = 0; s < 4; ++s) {
        int q  = pP * 4 + s;  int sP = q / NB,  pf = q % NB;     // reshape scramble
        int q2 = aP * 4 + s;  int sA = q2 / NB, af = q2 % NB;
        const unsigned char* brow = bins + (size_t)((bc * 4 + sP) * NB + pf) * T_LEN;
        const float*         arow = amps + (size_t)((bc * 4 + sA) * NB + af) * T_LEN;
        for (int t = tid; t < T_LEN; t += 256) {
            int k = (int)brow[t];
            hsum[tid * N_BINS + k] += arow[t];
            hcnt[tid * N_BINS + k] += 1.0f;
        }
    }
    __syncthreads();

    __shared__ float fsum[N_BINS], fcnt[N_BINS];
    if (tid < N_BINS) {                             // fixed-order serial reduce
        float s = 0.f, c = 0.f;
        for (int j = 0; j < 256; ++j) {
            s += hsum[j * N_BINS + tid];
            c += hcnt[j * N_BINS + tid];
        }
        fsum[tid] = s;
        fcnt[tid] = c;
    }
    __syncthreads();

    if (tid == 0) {
        float mean[N_BINS], tot = 0.f;
        for (int k = 0; k < N_BINS; ++k) {
            float m = fsum[k] / fmaxf(fcnt[k], 1.0f);
            mean[k] = m;
            tot += m;
        }
        tot = fmaxf(tot, 1e-10f);
        const float logK = logf((float)N_BINS);
        float acc = logK;
        for (int k = 0; k < N_BINS; ++k) {
            float p = mean[k] / tot;
            acc += p * logf(p + 1e-10f);
        }
        out[(bc * NB + pP) * NB + aP] = acc / logK;
    }
}

// ---------------------------------------------------------------------------
extern "C" void kernel_launch(void* const* d_in, const int* in_sizes, int n_in,
                              void* d_out, int out_size, void* d_ws, size_t ws_size,
                              hipStream_t stream) {
    const float* x    = (const float*)d_in[0];      // (4,8,4,4096)
    const float* phaK = (const float*)d_in[1];      // (10,385)
    const float* ampK = (const float*)d_in[2];      // (10,193)

    char* ws = (char*)d_ws;
    _Float16* phaR = (_Float16*)(ws + WS_PHA_R);
    _Float16* phaQ = (_Float16*)(ws + WS_PHA_Q);
    _Float16* ampR = (_Float16*)(ws + WS_AMP_R);
    _Float16* ampQ = (_Float16*)(ws + WS_AMP_Q);
    unsigned char* bins = (unsigned char*)(ws + WS_BINS);
    float* amps = (float*)(ws + WS_AMPS);

    (void)in_sizes; (void)n_in; (void)out_size; (void)ws_size;

    int prepThreads = (CH_PHA + CH_AMP) * 32 * 16;  // 10240
    pac_prep_banks<<<(prepThreads + 255) / 256, 256, 0, stream>>>(
        phaK, ampK, phaR, phaQ, ampR, ampQ);

    // grid: 32 time blocks x 128 signals; 256 threads = 8 waves per block.
    pac_filter_wmma<0><<<dim3(T_LEN / 128, N_SIG), 256, 0, stream>>>(
        x, phaR, phaQ, ampR, ampQ, bins, amps);
    pac_filter_wmma<1><<<dim3(T_LEN / 128, N_SIG), 256, 0, stream>>>(
        x, phaR, phaQ, ampR, ampQ, bins, amps);

    // one block per (a', p', b*8+c) output element.
    pac_mi<<<dim3(NB, NB, 32), 256, 0, stream>>>(bins, amps, (float*)d_out);
}
